// VectorQuantizer_61418032332822
// MI455X (gfx1250) — matrix operand order repaired and checked by reading, not compiled, not run on hardware
//
#include <hip/hip_runtime.h>
#include <hip/hip_bf16.h>
#include <cstdint>

typedef __attribute__((ext_vector_type(16))) _Float16 v16h;
typedef __attribute__((ext_vector_type(8)))  _Float16 v8h;
typedef __attribute__((ext_vector_type(8)))  float    v8f;

#define N_TOKENS 131072
#define EMB_DIM  128
#define NUM_EMB  1024
#define TILES    (N_TOKENS / 16)          // 8192
#define LDS_ROW_H 136                     // 128 halves + 8 pad -> 272B row stride (16B aligned, bank-skewed)

// ---------------------------------------------------------------------------
// Kernel A: L2-normalize the codebook into f16 (workspace) + zero accumulators
// ---------------------------------------------------------------------------
__global__ __launch_bounds__(32) void vq_norm_codebook(const float* __restrict__ weight,
                                                       _Float16* __restrict__ cnh,
                                                       float* __restrict__ acc) {
    int k = blockIdx.x;          // one wave per codebook row
    int l = threadIdx.x;         // 0..31
    if (k == 0) {                // zero mse (acc[0]) + probs (acc[64..1088))
        for (int i = l; i < 1088; i += 32) acc[i] = 0.0f;
    }
    const float* row = weight + (size_t)k * EMB_DIM;
    float v0 = row[l * 4 + 0], v1 = row[l * 4 + 1], v2 = row[l * 4 + 2], v3 = row[l * 4 + 3];
    float ss = v0 * v0 + v1 * v1 + v2 * v2 + v3 * v3;
    #pragma unroll
    for (int off = 16; off >= 1; off >>= 1) ss += __shfl_xor(ss, off, 32);
    float scale = 1.0f / fmaxf(sqrtf(ss), 1e-12f);
    _Float16* dst = cnh + (size_t)k * EMB_DIM + l * 4;
    dst[0] = (_Float16)(v0 * scale);
    dst[1] = (_Float16)(v1 * scale);
    dst[2] = (_Float16)(v2 * scale);
    dst[3] = (_Float16)(v3 * scale);
}

// ---------------------------------------------------------------------------
// Main kernel: WMMA similarity + softmax + argmax + gather + losses.
// Scores are 2*dot of unit vectors -> bounded in [-2,2], so exp() needs no
// max-shift: plain sum of exp is numerically safe and much cheaper.
// 256 threads = 8 waves; one 16-token tile per wave; full codebook in LDS.
// ---------------------------------------------------------------------------
__global__ __launch_bounds__(256) void vq_main(const float* __restrict__ z_e,
                                               const float* __restrict__ weight,
                                               const _Float16* __restrict__ cnh,
                                               float* __restrict__ out,
                                               float* __restrict__ probs_acc,
                                               float* __restrict__ mse_acc) {
    extern __shared__ char smem[];
    _Float16* cb       = (_Float16*)smem;                          // 1024 rows x 272B
    float*    probs_l  = (float*)(smem + NUM_EMB * LDS_ROW_H * 2); // 1024 f32
    int*      rowidx   = (int*)(smem + NUM_EMB * LDS_ROW_H * 2 + NUM_EMB * 4); // 8 waves x 16

    const int tid  = threadIdx.x;
    const int wave = tid >> 5;
    const int lane = tid & 31;
    const int col  = lane & 15;    // row-in-tile for A, column for B/C
    const int h    = lane >> 4;    // lane half

    // zero LDS probability histogram
    for (int i = tid; i < NUM_EMB; i += 256) probs_l[i] = 0.0f;

    // cooperative fill: f16 codebook -> padded LDS rows (16B chunks)
    for (int c = tid; c < NUM_EMB * (EMB_DIM / 8); c += 256) {
        int k   = c >> 4;            // 16 chunks of 8 halves per row
        int off = (c & 15) * 8;
        *(uint4*)(cb + k * LDS_ROW_H + off) = *(const uint4*)(cnh + (size_t)k * EMB_DIM + off);
    }
    __syncthreads();

    const int tile = blockIdx.x * 8 + wave;     // exactly TILES tiles total
    const int grow = tile * 16;

    // ---- load token tile row (grow+col), normalize, build 4 A fragments ----
    const float* zr = z_e + (size_t)(grow + col) * EMB_DIM;
    float zb[64];
    float ss = 0.0f;
    #pragma unroll
    for (int c2 = 0; c2 < 4; ++c2) {
        #pragma unroll
        for (int g = 0; g < 2; ++g) {           // A layout: e<8 -> d=32c+8h+e ; e>=8 -> +16
            int d0 = 32 * c2 + 8 * h + 16 * g;
            float4 p0 = *(const float4*)(zr + d0);
            float4 p1 = *(const float4*)(zr + d0 + 4);
            int e = c2 * 16 + g * 8;
            zb[e + 0] = p0.x; zb[e + 1] = p0.y; zb[e + 2] = p0.z; zb[e + 3] = p0.w;
            zb[e + 4] = p1.x; zb[e + 5] = p1.y; zb[e + 6] = p1.z; zb[e + 7] = p1.w;
        }
    }
    #pragma unroll
    for (int i = 0; i < 64; ++i) ss += zb[i] * zb[i];
    ss += __shfl_xor(ss, 16, 32);
    float zscale = 1.0f / fmaxf(sqrtf(ss), 1e-12f);

    v16h afrag[4];
    #pragma unroll
    for (int c2 = 0; c2 < 4; ++c2) {
        #pragma unroll
        for (int e = 0; e < 16; ++e)
            afrag[c2][e] = (_Float16)(zb[c2 * 16 + e] * zscale);
    }

    // one 16x16 tile of dot scores against codes [n0, n0+16)
    auto compute_tile = [&](int n0) -> v8f {
        v8f acc = {};
        #pragma unroll
        for (int c2 = 0; c2 < 4; ++c2) {
            const _Float16* bp = cb + (n0 + col) * LDS_ROW_H + 32 * c2 + 8 * h;
            v8h lo = *(const v8h*)(bp);
            v8h hi = *(const v8h*)(bp + 16);
            v16h bf;
            #pragma unroll
            for (int e = 0; e < 8; ++e) { bf[e] = lo[e]; bf[e + 8] = hi[e]; }
            acc = __builtin_amdgcn_wmma_f32_16x16x32_f16(false, afrag[c2], false, bf,
                                                         (short)0, acc, false, false);
        }
        return acc;
    };

    // ---- pass 1: sum of exp(2*dot) + argmax per C-slot (no max-shift needed:
    //      scores bounded in [-2,2] for unit vectors) ----
    float sm[8], best[8];
    int bidx[8];
    #pragma unroll
    for (int v = 0; v < 8; ++v) { sm[v] = 0.0f; best[v] = -3.4e38f; bidx[v] = 0; }

    for (int n0 = 0; n0 < NUM_EMB; n0 += 16) {
        v8f acc = compute_tile(n0);
        #pragma unroll
        for (int v = 0; v < 8; ++v) {
            float sc = 2.0f * acc[v];               // softmax(-dist) == softmax(2*dot)
            if (sc > best[v]) { best[v] = sc; bidx[v] = n0 + col; }
            sm[v] += __expf(sc);
        }
    }
    // reduce across the 16 columns (stay within lane half; slot v = row v or v+8)
    #pragma unroll
    for (int v = 0; v < 8; ++v) {
        #pragma unroll
        for (int off = 8; off >= 1; off >>= 1) {
            sm[v] += __shfl_xor(sm[v], off, 32);
            float ob = __shfl_xor(best[v], off, 32);
            int   oi = __shfl_xor(bidx[v], off, 32);
            if (ob > best[v] || (ob == best[v] && oi < bidx[v])) { best[v] = ob; bidx[v] = oi; }
        }
        sm[v] = 1.0f / sm[v];                        // now 1/Z for the slot's row
    }
    // publish per-row argmin indices for writer lanes
    if (lane == 0) {
        #pragma unroll
        for (int v = 0; v < 8; ++v) rowidx[wave * 16 + v] = bidx[v];
    }
    if (lane == 16) {
        #pragma unroll
        for (int v = 0; v < 8; ++v) rowidx[wave * 16 + 8 + v] = bidx[v];
    }
    __syncthreads();

    // ---- pass 2: replay WMMAs, accumulate column probability sums ----
    for (int n0 = 0; n0 < NUM_EMB; n0 += 16) {
        v8f acc = compute_tile(n0);
        float colsum = 0.0f;
        #pragma unroll
        for (int v = 0; v < 8; ++v)
            colsum += __expf(2.0f * acc[v]) * sm[v];
        colsum += __shfl_xor(colsum, 16, 32);        // add rows 8..15 (or 0..7) half
        if (lane < 16) atomicAdd(&probs_l[n0 + lane], colsum);
    }

    // ---- epilogue: gather z_q = weight[idx], write outputs, MSE partial ----
    if (lane < 16) {
        int g = grow + lane;
        int b = rowidx[wave * 16 + lane];
        const float* wr  = weight + (size_t)b * EMB_DIM;
        const float* zr2 = z_e + (size_t)g * EMB_DIM;
        float* outr = out + (size_t)g * EMB_DIM;
        float mse = 0.0f;
        #pragma unroll
        for (int d = 0; d < EMB_DIM; d += 4) {
            float4 w4 = *(const float4*)(wr + d);
            float4 z4 = *(const float4*)(zr2 + d);
            *(float4*)(outr + d) = w4;               // z_q_st forward value == z_q
            float dx = w4.x - z4.x, dy = w4.y - z4.y, dz = w4.z - z4.z, dw = w4.w - z4.w;
            mse += dx * dx + dy * dy + dz * dz + dw * dw;
        }
        atomicAdd(mse_acc, mse);
        out[(size_t)N_TOKENS * EMB_DIM + 3 + g] = (float)b;   // encoding index (as f32)
    }
    __syncthreads();
    // flush block-local probability histogram to global accumulator
    for (int i = tid; i < NUM_EMB; i += 256) atomicAdd(&probs_acc[i], probs_l[i]);
}

// ---------------------------------------------------------------------------
// Final reduction: entropy -> diversity loss, MSE -> vq loss, total loss
// ---------------------------------------------------------------------------
__global__ __launch_bounds__(256) void vq_final(const float* __restrict__ probs_acc,
                                                const float* __restrict__ mse_acc,
                                                float* __restrict__ out) {
    __shared__ float red[256];
    int t = threadIdx.x;
    float e = 0.0f;
    for (int i = t; i < NUM_EMB; i += 256) {
        float a = probs_acc[i] * (1.0f / (float)N_TOKENS);
        e += a * logf(a + 1e-10f);
    }
    red[t] = e;
    __syncthreads();
    for (int s = 128; s > 0; s >>= 1) { if (t < s) red[t] += red[t + s]; __syncthreads(); }
    if (t == 0) {
        float entropy = -red[0];
        float div = logf((float)NUM_EMB) - entropy;
        float msem = mse_acc[0] / ((float)N_TOKENS * (float)EMB_DIM);
        float vq = 2.0f * msem;                      // codebook + 1.0 * commitment (same fwd value)
        size_t off = (size_t)N_TOKENS * EMB_DIM;
        out[off + 0] = vq;
        out[off + 1] = div;
        out[off + 2] = vq + 0.1f * div;
    }
}

extern "C" void kernel_launch(void* const* d_in, const int* in_sizes, int n_in,
                              void* d_out, int out_size, void* d_ws, size_t ws_size,
                              hipStream_t stream) {
    const float* z_e    = (const float*)d_in[0];
    const float* weight = (const float*)d_in[1];
    float* out = (float*)d_out;

    float*     acc       = (float*)d_ws;                  // [0]=mse, [64..1088)=probs
    float*     mse_acc   = acc;
    float*     probs_acc = acc + 64;
    _Float16*  cnh       = (_Float16*)((char*)d_ws + 8192);

    (void)in_sizes; (void)n_in; (void)out_size; (void)ws_size;

    vq_norm_codebook<<<NUM_EMB, 32, 0, stream>>>(weight, cnh, acc);

    size_t shmem = (size_t)NUM_EMB * LDS_ROW_H * 2 + NUM_EMB * 4 + 8 * 16 * 4; // ~283 KB < 320 KB
    (void)hipFuncSetAttribute((const void*)vq_main,
                              hipFuncAttributeMaxDynamicSharedMemorySize, (int)shmem);
    vq_main<<<TILES / 8, 256, shmem, stream>>>(z_e, weight, cnh, out, probs_acc, mse_acc);

    vq_final<<<1, 256, 0, stream>>>(probs_acc, mse_acc, out);
}
